// CustomMultiHeadSelfAttention_60464549593623
// MI455X (gfx1250) — compile-verified, matrix operand order checked by reading, not run on hardware
//
#include <hip/hip_runtime.h>

// ---------------- types ----------------
typedef __attribute__((ext_vector_type(16))) __bf16        v16bf;
typedef __attribute__((ext_vector_type(8)))  float         v8f;
typedef __attribute__((ext_vector_type(8)))  unsigned int  v8u;

#define WMMA_BF16(A, B, C) \
  __builtin_amdgcn_wmma_f32_16x16x32_bf16(false, (A), false, (B), (short)0, (C), false, false)

#define N_BATCH 2
#define L_SEQ   2048
#define EMB     1024
#define H_HEADS 16
#define D_HEAD  64
#define M_TOT   (N_BATCH * L_SEQ)   // 4096

// ---------------- helpers ----------------
__device__ __forceinline__ unsigned short f2bf(float f) {
  unsigned int u = __float_as_uint(f);
  u += 0x7fffu + ((u >> 16) & 1u);          // round-to-nearest-even
  return (unsigned short)(u >> 16);
}

// Load a 16x32 bf16 A-fragment (or mirrored B-fragment: lane = output column,
// reading "row n of W" = column n of B) from row-major memory.
// Layout (ISA 7.12.2, 16-bit A 16x32): lanes [0,16): khalf=0; lanes [16,32):
// khalf=8. Elements [0..7] = k in [khalf, khalf+8), elements [8..15] =
// k in [16+khalf, 16+khalf+8). Two 16-byte loads per lane.
__device__ __forceinline__ v16bf load_frag_g(const unsigned short* __restrict__ p0, int ld) {
  const int lane = threadIdx.x & 31;
  const unsigned short* p = p0 + (lane & 15) * ld + ((lane >> 4) << 3);
  uint4 lo = *(const uint4*)p;
  uint4 hi = *(const uint4*)(p + 16);
  v8u u = { lo.x, lo.y, lo.z, lo.w, hi.x, hi.y, hi.z, hi.w };
  return __builtin_bit_cast(v16bf, u);
}

// ---------------- kernel 1: fp32 -> bf16 cast ----------------
__global__ void cast_f32_bf16(const float* __restrict__ in,
                              unsigned short* __restrict__ out, int n) {
  int i = blockIdx.x * blockDim.x + threadIdx.x;
  if (i < n) out[i] = f2bf(in[i]);
}

// ---------------- kernel 2: QKV projection + fused per-head LayerNorm -------
// grid = (M_TOT/128, H, 3[q,k,v]); block = 128 (4 waves); each wave computes
// a 32x64 strip of (x @ W.T) covering one full head (two 16-row subtiles
// sharing the B fragments) -> per-head LayerNorm done fully in-register.
__global__ void __launch_bounds__(128) proj_qkv_ln(
    const unsigned short* __restrict__ xb,
    const unsigned short* __restrict__ wqb,
    const unsigned short* __restrict__ wkb,
    const unsigned short* __restrict__ wvb,
    const float* __restrict__ gq, const float* __restrict__ bq,
    const float* __restrict__ gk, const float* __restrict__ bk,
    unsigned short* __restrict__ Qb,   // [N,H,L,D] bf16 (LayerNormed)
    unsigned short* __restrict__ Kb,   // [N,H,L,D] bf16 (LayerNormed)
    unsigned short* __restrict__ Vt)   // [N,H,D,L] bf16 (transposed)
{
  const int tid   = threadIdx.x;
  const int wv    = tid >> 5;
  const int lane  = tid & 31;
  const int nn    = lane & 15;
  const int mbase = (lane >> 4) << 3;
  const int bm    = blockIdx.x * 128 + wv * 32;  // first of 32 rows for wave
  const int h     = blockIdx.y;
  const int mode  = blockIdx.z;                  // 0=Q 1=K 2=V

  const unsigned short* wb = (mode == 0) ? wqb : (mode == 1) ? wkb : wvb;

  v8f acc[2][4];
#pragma unroll
  for (int s = 0; s < 2; ++s)
#pragma unroll
    for (int f = 0; f < 4; ++f) acc[s][f] = v8f{};

  for (int kk = 0; kk < EMB; kk += 32) {
    v16bf a0 = load_frag_g(xb + (bm +  0) * EMB + kk, EMB);
    v16bf a1 = load_frag_g(xb + (bm + 16) * EMB + kk, EMB);
#pragma unroll
    for (int f = 0; f < 4; ++f) {
      v16bf bfr = load_frag_g(wb + (h * D_HEAD + f * 16) * EMB + kk, EMB);
      acc[0][f] = WMMA_BF16(a0, bfr, acc[0][f]);
      acc[1][f] = WMMA_BF16(a1, bfr, acc[1][f]);
    }
  }

  if (mode == 2) {
    // V: no LN; store transposed [N,H,D,L]
#pragma unroll
    for (int s = 0; s < 2; ++s)
#pragma unroll
      for (int r = 0; r < 8; ++r) {
        const int m  = bm + s * 16 + mbase + r;
        const int nb = m >> 11;             // / L_SEQ
        const int l  = m & (L_SEQ - 1);
#pragma unroll
        for (int f = 0; f < 4; ++f)
          Vt[(((nb * H_HEADS + h) * D_HEAD) + f * 16 + nn) * L_SEQ + l] =
              f2bf(acc[s][f][r]);
      }
    return;
  }

  const float* g = (mode == 0) ? gq : gk;
  const float* b = (mode == 0) ? bq : bk;
  float gv[4], bv[4];
#pragma unroll
  for (int f = 0; f < 4; ++f) { gv[f] = g[f * 16 + nn]; bv[f] = b[f * 16 + nn]; }

  unsigned short* outp = (mode == 0) ? Qb : Kb;
#pragma unroll
  for (int s = 0; s < 2; ++s)
#pragma unroll
    for (int r = 0; r < 8; ++r) {
      // mean over D=64: 4 frags per lane, reduce across 16-lane half
      float sm = acc[s][0][r] + acc[s][1][r] + acc[s][2][r] + acc[s][3][r];
      sm += __shfl_xor(sm, 1, 16); sm += __shfl_xor(sm, 2, 16);
      sm += __shfl_xor(sm, 4, 16); sm += __shfl_xor(sm, 8, 16);
      const float mean = sm * 0.015625f;
      float d0 = acc[s][0][r] - mean, d1 = acc[s][1][r] - mean,
            d2 = acc[s][2][r] - mean, d3 = acc[s][3][r] - mean;
      float sq = d0 * d0 + d1 * d1 + d2 * d2 + d3 * d3;
      sq += __shfl_xor(sq, 1, 16); sq += __shfl_xor(sq, 2, 16);
      sq += __shfl_xor(sq, 4, 16); sq += __shfl_xor(sq, 8, 16);
      const float rstd = rsqrtf(sq * 0.015625f + 1e-5f);

      const int m  = bm + s * 16 + mbase + r;
      const int nb = m >> 11;
      const int l  = m & (L_SEQ - 1);
      unsigned short* dst = outp + (((nb * H_HEADS + h) * L_SEQ) + l) * D_HEAD;
      dst[0 * 16 + nn] = f2bf(d0 * rstd * gv[0] + bv[0]);
      dst[1 * 16 + nn] = f2bf(d1 * rstd * gv[1] + bv[1]);
      dst[2 * 16 + nn] = f2bf(d2 * rstd * gv[2] + bv[2]);
      dst[3 * 16 + nn] = f2bf(d3 * rstd * gv[3] + bv[3]);
    }
}

// ---------------- kernel 3: causal flash attention ----------------
// grid = (L/64, H, N); block = 128 (4 independent waves, 16 query rows each).
__global__ void __launch_bounds__(128) attn_fwd(
    const unsigned short* __restrict__ Qb,
    const unsigned short* __restrict__ Kb,
    const unsigned short* __restrict__ Vt,
    unsigned short* __restrict__ Ob)     // [N,L,EMB] bf16
{
  __shared__ __align__(16) float          S[4][16][32];
  __shared__ __align__(16) unsigned short P[4][16][32];
  __shared__ float                        SCL[4][16];

  const int tid   = threadIdx.x;
  const int wv    = tid >> 5;
  const int lane  = tid & 31;
  const int nn    = lane & 15;
  const int mbase = (lane >> 4) << 3;
  const int h     = blockIdx.y;
  const int nb    = blockIdx.z;
  const int qbase = blockIdx.x * 64 + wv * 16;

  const unsigned short* Qh = Qb + (size_t)((nb * H_HEADS + h) * L_SEQ) * D_HEAD;
  const unsigned short* Kh = Kb + (size_t)((nb * H_HEADS + h) * L_SEQ) * D_HEAD;
  const unsigned short* Vh = Vt + (size_t)((nb * H_HEADS + h) * D_HEAD) * L_SEQ;

  const v16bf q0 = load_frag_g(Qh + qbase * D_HEAD + 0,  D_HEAD);
  const v16bf q1 = load_frag_g(Qh + qbase * D_HEAD + 32, D_HEAD);

  v8f o[4];
#pragma unroll
  for (int f = 0; f < 4; ++f) o[f] = v8f{};
  float mrun = -3.0e38f, lrun = 0.f;   // valid in lanes 0..15 (row = lane)

  const int nchw = (qbase + 16 + 31) >> 5;   // causal: keys < qbase+16
  for (int ch = 0; ch < nchw; ++ch) {
    const int kb = ch * 32;
    // S = Q * K^T over D=64 (two k-steps) for two 16-key subtiles
    v16bf kA0 = load_frag_g(Kh + (kb +  0) * D_HEAD + 0,  D_HEAD);
    v16bf kA1 = load_frag_g(Kh + (kb +  0) * D_HEAD + 32, D_HEAD);
    v16bf kB0 = load_frag_g(Kh + (kb + 16) * D_HEAD + 0,  D_HEAD);
    v16bf kB1 = load_frag_g(Kh + (kb + 16) * D_HEAD + 32, D_HEAD);
    v8f z{};
    v8f c0 = WMMA_BF16(q0, kA0, z);  c0 = WMMA_BF16(q1, kA1, c0);
    v8f c1 = WMMA_BF16(q0, kB0, z);  c1 = WMMA_BF16(q1, kB1, c1);

    // causal mask + spill scores to LDS (row-major 16x32)
#pragma unroll
    for (int r = 0; r < 8; ++r) {
      const int m = qbase + mbase + r;
      if (kb + nn > m)      c0[r] = -1e30f;
      if (kb + 16 + nn > m) c1[r] = -1e30f;
      S[wv][mbase + r][nn]      = c0[r];
      S[wv][mbase + r][16 + nn] = c1[r];
    }
    asm volatile("s_wait_dscnt 0" ::: "memory");

    // online softmax: lanes 0..15 each own one row
    if (lane < 16) {
      float mx = mrun;
#pragma unroll
      for (int j = 0; j < 32; ++j) mx = fmaxf(mx, S[wv][lane][j]);
      const float sc = __expf(mrun - mx);
      float sum = 0.f;
#pragma unroll
      for (int j = 0; j < 32; ++j) {
        const float p = __expf(S[wv][lane][j] - mx);
        sum += p;
        P[wv][lane][j] = f2bf(p);
      }
      lrun = lrun * sc + sum;
      mrun = mx;
      SCL[wv][lane] = sc;
    }
    asm volatile("s_wait_dscnt 0" ::: "memory");

    // rescale O accumulators by per-row correction factor
#pragma unroll
    for (int r = 0; r < 8; ++r) {
      const float s = SCL[wv][mbase + r];
#pragma unroll
      for (int f = 0; f < 4; ++f) o[f][r] *= s;
    }

    // reload P as bf16 A-fragment from LDS (two ds_load_b128)
    {
      const int kh = (lane >> 4) << 3;
      uint4 plo = *(const uint4*)&P[wv][nn][kh];
      uint4 phi = *(const uint4*)&P[wv][nn][kh + 16];
      v8u pu = { plo.x, plo.y, plo.z, plo.w, phi.x, phi.y, phi.z, phi.w };
      v16bf pf = __builtin_bit_cast(v16bf, pu);
#pragma unroll
      for (int f = 0; f < 4; ++f) {
        v16bf vb = load_frag_g(Vh + (f * 16) * L_SEQ + kb, L_SEQ);  // V^T rows
        o[f] = WMMA_BF16(pf, vb, o[f]);
      }
    }
  }

  // normalize and store [N,L,EMB] bf16
  if (lane < 16) SCL[wv][lane] = 1.0f / lrun;
  asm volatile("s_wait_dscnt 0" ::: "memory");
#pragma unroll
  for (int r = 0; r < 8; ++r) {
    const float inv = SCL[wv][mbase + r];
    const int l = qbase + mbase + r;
    unsigned short* row = Ob + (size_t)(nb * L_SEQ + l) * EMB + h * D_HEAD;
#pragma unroll
    for (int f = 0; f < 4; ++f) row[f * 16 + nn] = f2bf(o[f][r] * inv);
  }
}

// ---------------- kernel 4: output projection + bias ----------------
// grid = (M_TOT/128, EMB/64); block = 128; wave computes 32x64 fp32 tile
// (two 16-row subtiles sharing B fragments).
__global__ void __launch_bounds__(128) out_proj(
    const unsigned short* __restrict__ Ab,   // [M_TOT, EMB] bf16
    const unsigned short* __restrict__ Wob,  // [EMB, EMB] bf16 row-major
    const float* __restrict__ bo,
    float* __restrict__ out)
{
  const int tid   = threadIdx.x;
  const int wv    = tid >> 5;
  const int lane  = tid & 31;
  const int nn    = lane & 15;
  const int mbase = (lane >> 4) << 3;
  const int bm    = blockIdx.x * 128 + wv * 32;
  const int bc    = blockIdx.y * 64;

  v8f acc[2][4];
#pragma unroll
  for (int s = 0; s < 2; ++s)
#pragma unroll
    for (int f = 0; f < 4; ++f) acc[s][f] = v8f{};

  for (int kk = 0; kk < EMB; kk += 32) {
    v16bf a0 = load_frag_g(Ab + (bm +  0) * EMB + kk, EMB);
    v16bf a1 = load_frag_g(Ab + (bm + 16) * EMB + kk, EMB);
#pragma unroll
    for (int f = 0; f < 4; ++f) {
      v16bf bfr = load_frag_g(Wob + (bc + f * 16) * EMB + kk, EMB);
      acc[0][f] = WMMA_BF16(a0, bfr, acc[0][f]);
      acc[1][f] = WMMA_BF16(a1, bfr, acc[1][f]);
    }
  }
#pragma unroll
  for (int s = 0; s < 2; ++s)
#pragma unroll
    for (int r = 0; r < 8; ++r) {
      const int m = bm + s * 16 + mbase + r;
      float* row = out + (size_t)m * EMB + bc;
#pragma unroll
      for (int f = 0; f < 4; ++f)
        row[f * 16 + nn] = acc[s][f][r] + bo[bc + f * 16 + nn];
    }
}

// ---------------- host launch ----------------
extern "C" void kernel_launch(void* const* d_in, const int* in_sizes, int n_in,
                              void* d_out, int out_size, void* d_ws, size_t ws_size,
                              hipStream_t stream) {
  const float* x  = (const float*)d_in[0];
  // d_in[1] = mask (unused by reference)
  const float* Wq = (const float*)d_in[2];
  const float* Wk = (const float*)d_in[3];
  const float* Wv = (const float*)d_in[4];
  const float* gq = (const float*)d_in[5];
  const float* bq = (const float*)d_in[6];
  const float* gk = (const float*)d_in[7];
  const float* bk = (const float*)d_in[8];
  const float* Wo = (const float*)d_in[9];
  const float* bo = (const float*)d_in[10];

  char* ws = (char*)d_ws;
  // workspace layout (48 MiB total)
  unsigned short* xb  = (unsigned short*)(ws + (size_t)0);          // 8 MiB
  unsigned short* wqb = (unsigned short*)(ws + ((size_t)8  << 20)); // 2 MiB
  unsigned short* wkb = (unsigned short*)(ws + ((size_t)10 << 20));
  unsigned short* wvb = (unsigned short*)(ws + ((size_t)12 << 20));
  unsigned short* wob = (unsigned short*)(ws + ((size_t)14 << 20));
  unsigned short* Qb  = (unsigned short*)(ws + ((size_t)16 << 20)); // 8 MiB
  unsigned short* Kb  = (unsigned short*)(ws + ((size_t)24 << 20));
  unsigned short* Vt  = (unsigned short*)(ws + ((size_t)32 << 20));
  unsigned short* Ob  = (unsigned short*)(ws + ((size_t)40 << 20));

  const int nx = N_BATCH * L_SEQ * EMB;   // 4194304
  const int nw = EMB * EMB;               // 1048576
  cast_f32_bf16<<<(nx + 255) / 256, 256, 0, stream>>>(x,  xb,  nx);
  cast_f32_bf16<<<(nw + 255) / 256, 256, 0, stream>>>(Wq, wqb, nw);
  cast_f32_bf16<<<(nw + 255) / 256, 256, 0, stream>>>(Wk, wkb, nw);
  cast_f32_bf16<<<(nw + 255) / 256, 256, 0, stream>>>(Wv, wvb, nw);
  cast_f32_bf16<<<(nw + 255) / 256, 256, 0, stream>>>(Wo, wob, nw);

  proj_qkv_ln<<<dim3(M_TOT / 128, H_HEADS, 3), 128, 0, stream>>>(
      xb, wqb, wkb, wvb, gq, bq, gk, bk, Qb, Kb, Vt);

  attn_fwd<<<dim3(L_SEQ / 64, H_HEADS, N_BATCH), 128, 0, stream>>>(Qb, Kb, Vt, Ob);

  out_proj<<<dim3(M_TOT / 128, EMB / 64), 128, 0, stream>>>(Ob, wob, bo, (float*)d_out);
}
